// Block_39874476376768
// MI455X (gfx1250) — matrix operand order repaired and checked by reading, not compiled, not run on hardware
//
#include <hip/hip_runtime.h>
#include <math.h>
#include <stdint.h>

// ---------------------------------------------------------------------------
// Types for CDNA5 WMMA (wave32). 16x16x32 bf16 -> f32 accumulate.
// ---------------------------------------------------------------------------
typedef __attribute__((ext_vector_type(16))) __bf16 v16bf;
typedef __attribute__((ext_vector_type(8)))  __bf16 v8bf;
typedef __attribute__((ext_vector_type(8)))  float  v8f;

#define WMMA_BF16(a, b, c) \
  __builtin_amdgcn_wmma_f32_16x16x32_bf16(false, (a), false, (b), (short)0, (c), false, false)

static constexpr int BB = 2;      // batch
static constexpr int S  = 2048;   // sequence
static constexpr int E  = 1024;   // embed
static constexpr int F  = 4096;   // ffn
static constexpr int H  = 16;     // heads
static constexpr int D  = 64;     // head dim
static constexpr int M  = BB * S; // 4096 rows total
static constexpr float NEGMASK = -10000.0f;
static constexpr float LN_EPS  = 1e-5f;

// ---------------------------------------------------------------------------
// CDNA5 async global->LDS copy (ASYNCcnt-tracked DMA into LDS).
// LDS aperture: generic addr[31:0] is the LDS byte offset (ISA 10.2).
// ---------------------------------------------------------------------------
// #define NO_ASYNC_LDS 1   // <- flip if the assembler rejects the mnemonic

__device__ __forceinline__ void async_ld_b128(void* lds_ptr, const void* gptr) {
#ifndef NO_ASYNC_LDS
  uint32_t lds_byte = (uint32_t)(uintptr_t)lds_ptr;
  asm volatile("global_load_async_to_lds_b128 %0, %1, off"
               :
               : "v"(lds_byte), "v"(gptr)
               : "memory");
#else
  *(v8bf*)lds_ptr = *(const v8bf*)gptr;
#endif
}

__device__ __forceinline__ void wait_async0() {
#ifndef NO_ASYNC_LDS
#if __has_builtin(__builtin_amdgcn_s_wait_asynccnt)
  __builtin_amdgcn_s_wait_asynccnt(0);
#else
  asm volatile("s_wait_asynccnt 0" ::: "memory");
#endif
#endif
}

// 16-element bf16 fragment load from 16B-aligned memory (two b128 LDS loads).
__device__ __forceinline__ v16bf ld_frag(const __bf16* p) {
  v8bf lo = *(const v8bf*)p;
  v8bf hi = *(const v8bf*)(p + 8);
  return __builtin_shufflevector(lo, hi, 0, 1, 2, 3, 4, 5, 6, 7,
                                 8, 9, 10, 11, 12, 13, 14, 15);
}

// ---------------------------------------------------------------------------
// Weight downconvert + transpose: src fp32 [N,K] row-major -> dst bf16 [K,N].
// 32x32 LDS-tiled so both the fp32 reads and bf16 writes are coalesced.
// ---------------------------------------------------------------------------
__global__ __launch_bounds__(256) void wt_kernel(const float* __restrict__ src,
                                                 __bf16* __restrict__ dst,
                                                 int N, int K) {
  __shared__ float tile[32][33];
  const int ktiles = K / 32;
  const int k0 = (blockIdx.x % ktiles) * 32;
  const int n0 = (blockIdx.x / ktiles) * 32;
  const int r = threadIdx.x >> 5;   // 0..7
  const int c = threadIdx.x & 31;   // 0..31
#pragma unroll
  for (int i = 0; i < 4; ++i)
    tile[r + i * 8][c] = src[(size_t)(n0 + r + i * 8) * K + k0 + c];
  __syncthreads();
#pragma unroll
  for (int i = 0; i < 4; ++i)
    dst[(size_t)(k0 + r + i * 8) * N + n0 + c] = (__bf16)tile[c][r + i * 8];
}

// ---------------------------------------------------------------------------
// Fused (optional add-residual) + LayerNorm. One block per row, 256 threads,
// 4 elements each (E = 1024). Writes fp32 residual (optional) + bf16 normed.
// ---------------------------------------------------------------------------
__global__ __launch_bounds__(256) void add_ln_kernel(const float* __restrict__ x,
                                                     const float* __restrict__ r,
                                                     const float* __restrict__ w,
                                                     const float* __restrict__ b,
                                                     float* __restrict__ resid_out,
                                                     __bf16* __restrict__ h_out) {
  __shared__ float red[256];
  const int row = blockIdx.x;
  const int t   = threadIdx.x;
  const float* xr = x + (size_t)row * E;
  const float* rr = r ? r + (size_t)row * E : nullptr;

  float v[4];
  float s = 0.f;
#pragma unroll
  for (int i = 0; i < 4; ++i) {
    int idx = i * 256 + t;
    float val = xr[idx] + (rr ? rr[idx] : 0.f);
    v[i] = val;
    s += val;
    if (resid_out) resid_out[(size_t)row * E + idx] = val;
  }
  red[t] = s;
  __syncthreads();
  for (int off = 128; off > 0; off >>= 1) {
    if (t < off) red[t] += red[t + off];
    __syncthreads();
  }
  float mu = red[0] * (1.0f / E);
  __syncthreads();

  float s2 = 0.f;
#pragma unroll
  for (int i = 0; i < 4; ++i) {
    float d = v[i] - mu;
    s2 += d * d;
  }
  red[t] = s2;
  __syncthreads();
  for (int off = 128; off > 0; off >>= 1) {
    if (t < off) red[t] += red[t + off];
    __syncthreads();
  }
  float rs = rsqrtf(red[0] * (1.0f / E) + LN_EPS);

#pragma unroll
  for (int i = 0; i < 4; ++i) {
    int idx = i * 256 + t;
    float hn = (v[i] - mu) * rs * w[idx] + b[idx];
    h_out[(size_t)row * E + idx] = (__bf16)hn;
  }
}

// ---------------------------------------------------------------------------
// Tiled WMMA GEMM: C[M,N] = A[M,K] (bf16) x Bt[K,N] (bf16, pre-transposed
// weight) + bias, with templated epilogues.
//   EPI 0: QKV scatter -> q[B,H,S,D], kT[B,H,D,S], v[B,H,S,D] (bf16)
//   EPI 1: fp32 out = acc + bias + resid_in  (out projection -> resid2)
//   EPI 2: bf16 out = gelu(acc + bias)       (fc1)
//   EPI 3: fp32 out = acc + bias             (fc2 -> y)
// Block tile 128x128, 8 waves (4 row-groups x 2 col-groups), each wave 2x4
// WMMA 16x16 tiles, K-step 32, double-buffered async global->LDS staging.
// LDS rows padded (A: 40, B: 136 elems) so fragment reads spread banks.
// ---------------------------------------------------------------------------
static constexpr int PADA = 40;   // 80B row stride  -> banks rlo*20 mod 64
static constexpr int PADB = 136;  // 272B row stride -> banks lane*4 mod 64

template <int EPI>
__global__ __launch_bounds__(256) void gemm_kernel(const __bf16* __restrict__ A,
                                                   const __bf16* __restrict__ Bt,
                                                   const float* __restrict__ bias,
                                                   int N, int K,
                                                   float* __restrict__ outF,
                                                   __bf16* __restrict__ outB,
                                                   const float* __restrict__ resid_in,
                                                   __bf16* __restrict__ qp,
                                                   __bf16* __restrict__ kTp,
                                                   __bf16* __restrict__ vp) {
  __shared__ __align__(32) __bf16 lA[2][128 * PADA];  // [buf][row*PADA+col]
  __shared__ __align__(32) __bf16 lB[2][32 * PADB];   // [buf][row*PADB+col]

  const int t    = threadIdx.x;
  const int lane = t & 31;
  const int wid  = t >> 5;
  const int wr   = wid & 3;   // wave row group (32 rows each)
  const int wc   = wid >> 2;  // wave col group (64 cols each)
  const int m0   = blockIdx.y * 128;
  const int n0   = blockIdx.x * 128;
  const int rlo  = lane & 15;
  const int khalf = (lane >> 4) * 16;

  // Per-thread staging: 2 x 16B chunks of A-tile + 2 x 16B chunks of B-tile.
  // A tile 128x32 (4 chunks/row), B tile 32x128 (16 chunks/row).
  auto stage = [&](int buf, int k0) {
#pragma unroll
    for (int i = 0; i < 2; ++i) {
      int c = t * 2 + i;  // 0..511
      async_ld_b128(&lA[buf][(c >> 2) * PADA + (c & 3) * 8],
                    &A[(size_t)(m0 + (c >> 2)) * K + k0 + (c & 3) * 8]);
      async_ld_b128(&lB[buf][(c >> 4) * PADB + (c & 15) * 8],
                    &Bt[(size_t)(k0 + (c >> 4)) * N + n0 + (c & 15) * 8]);
    }
  };

  v8f acc[2][4] = {};

  stage(0, 0);
  wait_async0();
  __syncthreads();

  int cur = 0;
  for (int k0 = 0; k0 < K; k0 += 32) {
    int nxt = cur ^ 1;
    if (k0 + 32 < K) stage(nxt, k0 + 32);  // fill other buffer while computing

    const __bf16* pA = lA[cur];
    const __bf16* pB = lB[cur];
    v16bf af[2], bfr[4];
#pragma unroll
    for (int ti = 0; ti < 2; ++ti)
      af[ti] = ld_frag(&pA[(wr * 32 + ti * 16 + rlo) * PADA + khalf]);
#pragma unroll
    for (int tj = 0; tj < 4; ++tj)
#pragma unroll
      for (int s = 0; s < 16; ++s)
        bfr[tj][s] = pB[(khalf + s) * PADB + wc * 64 + tj * 16 + rlo];
#pragma unroll
    for (int ti = 0; ti < 2; ++ti)
#pragma unroll
      for (int tj = 0; tj < 4; ++tj)
        acc[ti][tj] = WMMA_BF16(af[ti], bfr[tj], acc[ti][tj]);

    wait_async0();     // my async writes into nxt are done
    __syncthreads();   // everyone finished reading cur & staging nxt
    cur = nxt;
  }

  // Epilogue. C layout: element (row = v + 8*(lane>=16), col = lane&15).
  const int hi = lane >> 4;
#pragma unroll
  for (int ti = 0; ti < 2; ++ti) {
#pragma unroll
    for (int tj = 0; tj < 4; ++tj) {
      int n = n0 + wc * 64 + tj * 16 + rlo;
      float bn = bias[n];
#pragma unroll
      for (int vv = 0; vv < 8; ++vv) {
        int m = m0 + wr * 32 + ti * 16 + vv + 8 * hi;
        float val = acc[ti][tj][vv] + bn;
        if constexpr (EPI == 0) {
          int bidx = m >> 11;          // m / S
          int sidx = m & (S - 1);      // m % S
          int part = n >> 10;          // n / E
          int nn   = n & (E - 1);
          int hh   = nn >> 6;
          int dd   = nn & 63;
          size_t bh = (size_t)(bidx * H + hh);
          if (part == 0)
            qp[(bh * S + sidx) * D + dd] = (__bf16)val;
          else if (part == 1)
            kTp[(bh * D + dd) * S + sidx] = (__bf16)val;
          else
            vp[(bh * S + sidx) * D + dd] = (__bf16)val;
        } else if constexpr (EPI == 1) {
          size_t o = (size_t)m * N + n;
          outF[o] = val + resid_in[o];
        } else if constexpr (EPI == 2) {
          float g = 0.5f * val * (1.0f + erff(val * 0.70710678118654752f));
          outB[(size_t)m * N + n] = (__bf16)g;
        } else {
          outF[(size_t)m * N + n] = val;
        }
      }
    }
  }
}

// ---------------------------------------------------------------------------
// Flash attention (causal). grid = (S/128, B*H); 8 waves/block; each wave owns
// a 16-row Q tile. The block cooperatively async-stages each 32-column K
// (64x32, kT layout) and V (32x64) chunk into double-buffered LDS shared by
// all 8 waves (8x less global traffic than per-wave streaming).
//   scores = (Q Kt) via 4 WMMAs -> per-wave LDS transpose ->
//   online softmax (row = lane&15, half = lane>>4 == the 16x32 A-fragment
//   layout, so P stays in registers) -> O += P V via 4 WMMAs (fp32 16x64).
// ---------------------------------------------------------------------------
static constexpr int PADK = 40;  // K-chunk rows (d) padded: 80B stride
static constexpr int PADV = 72;  // V-chunk rows (kj) padded: 144B stride

__global__ __launch_bounds__(256) void attn_kernel(const __bf16* __restrict__ q,
                                                   const __bf16* __restrict__ kT,
                                                   const __bf16* __restrict__ v,
                                                   __bf16* __restrict__ ctx) {
  __shared__ float sS[8][16 * 32];
  __shared__ __align__(32) __bf16 lK[2][64 * PADK];  // [buf][d*PADK + col]
  __shared__ __align__(32) __bf16 lV[2][32 * PADV];  // [buf][kj*PADV + dcol]

  const int t    = threadIdx.x;
  const int lane = t & 31;
  const int wid  = t >> 5;
  const int rlo  = lane & 15;
  const int hi   = lane >> 4;
  const int qblk  = blockIdx.x * 128;
  const int qbase = qblk + wid * 16;
  const int bh    = blockIdx.y;  // b*H + h

  const __bf16* qh = q  + (size_t)bh * S * D;
  const __bf16* kh = kT + (size_t)bh * D * S;
  const __bf16* vh = v  + (size_t)bh * S * D;
  float* smem = sS[wid];

  // Stage one 32-col chunk: K 64x32 (1 chunk/thread), V 32x64 (1 chunk/thread)
  auto stage = [&](int buf, int kc) {
    int ck = t;                       // 0..255 : K chunk, 4 per d-row
    async_ld_b128(&lK[buf][(ck >> 2) * PADK + (ck & 3) * 8],
                  &kh[(size_t)(ck >> 2) * S + kc + (ck & 3) * 8]);
    int cv = t;                       // 0..255 : V chunk, 8 per kj-row
    async_ld_b128(&lV[buf][(cv >> 3) * PADV + (cv & 7) * 8],
                  &vh[(size_t)(kc + (cv >> 3)) * D + (cv & 7) * 8]);
  };

  // Q fragments: 16 rows x 64 d = two 16x32 A-fragments.
  v16bf qf0 = *(const v16bf*)&qh[(size_t)(qbase + rlo) * D + hi * 16];
  v16bf qf1 = *(const v16bf*)&qh[(size_t)(qbase + rlo) * D + 32 + hi * 16];

  v8f O[4] = {};
  float mrow = -1e30f, lrow = 0.f;
  const float scale = 0.125f;  // 1/sqrt(64)
  const int qi   = qbase + rlo;
  const int kmax = qblk + 127;  // block-uniform causal bound

  stage(0, 0);
  wait_async0();
  __syncthreads();

  int cur = 0;
  for (int kc = 0; kc <= kmax; kc += 32) {
    int nxt = cur ^ 1;
    if (kc + 32 <= kmax) stage(nxt, kc + 32);

    if (kc <= qbase + 15) {  // wave-uniform: this wave needs this chunk
      const __bf16* pK = lK[cur];
      const __bf16* pV = lV[cur];
      // ---- scores: two 16x16 C tiles (cols kc..kc+31) ----
#pragma unroll
      for (int jt = 0; jt < 2; ++jt) {
        v16bf kf0, kf1;
#pragma unroll
        for (int s = 0; s < 16; ++s) {
          kf0[s] = pK[(size_t)(hi * 16 + s)      * PADK + jt * 16 + rlo];
          kf1[s] = pK[(size_t)(32 + hi * 16 + s) * PADK + jt * 16 + rlo];
        }
        v8f c = {};
        c = WMMA_BF16(qf0, kf0, c);
        c = WMMA_BF16(qf1, kf1, c);
#pragma unroll
        for (int vv = 0; vv < 8; ++vv)
          smem[(vv + 8 * hi) * 32 + jt * 16 + rlo] = c[vv];
      }
      // per-wave LDS is in-order within a wave: no barrier needed.

      // ---- online softmax: this lane owns row rlo, K half hi ----
      float p[16];
      float cmax = -1e30f;
#pragma unroll
      for (int j = 0; j < 16; ++j) {
        int col = kc + hi * 16 + j;
        float x = smem[rlo * 32 + hi * 16 + j] * scale;
        if (col > qi) x += NEGMASK;
        p[j] = x;
        cmax = fmaxf(cmax, x);
      }
      cmax = fmaxf(cmax, __shfl_xor(cmax, 16, 32));
      float newm  = fmaxf(mrow, cmax);
      float alpha = __expf(mrow - newm);
      float csum = 0.f;
      v16bf pf;
#pragma unroll
      for (int j = 0; j < 16; ++j) {
        float e = __expf(p[j] - newm);
        csum += e;
        pf[j] = (__bf16)e;  // exactly the 16x32 A-fragment layout
      }
      csum += __shfl_xor(csum, 16, 32);
      lrow = lrow * alpha + csum;
      mrow = newm;

      float av[8];
#pragma unroll
      for (int vv = 0; vv < 8; ++vv) av[vv] = __shfl(alpha, vv + 8 * hi, 32);

      // ---- O = O*alpha + P @ V  (four 16-col d groups) ----
#pragma unroll
      for (int dt = 0; dt < 4; ++dt) {
#pragma unroll
        for (int vv = 0; vv < 8; ++vv) O[dt][vv] *= av[vv];
        v16bf vf;
#pragma unroll
        for (int s = 0; s < 16; ++s)
          vf[s] = pV[(size_t)(hi * 16 + s) * PADV + dt * 16 + rlo];
        O[dt] = WMMA_BF16(pf, vf, O[dt]);
      }
    }

    wait_async0();     // my async writes into nxt are done
    __syncthreads();   // everyone finished reading cur & staging nxt
    cur = nxt;
  }

  float lv[8];
#pragma unroll
  for (int vv = 0; vv < 8; ++vv) {
    float lr = __shfl(lrow, vv + 8 * hi, 32);
    lv[vv] = 1.0f / lr;
  }
  const int bidx = bh >> 4, hh = bh & 15;
#pragma unroll
  for (int dt = 0; dt < 4; ++dt)
#pragma unroll
    for (int vv = 0; vv < 8; ++vv) {
      int mrowg = qbase + vv + 8 * hi;
      ctx[((size_t)(bidx * S + mrowg)) * E + hh * D + dt * 16 + rlo] =
          (__bf16)(O[dt][vv] * lv[vv]);
    }
}

// ---------------------------------------------------------------------------
// Host-side orchestration.
// ---------------------------------------------------------------------------
extern "C" void kernel_launch(void* const* d_in, const int* in_sizes, int n_in,
                              void* d_out, int out_size, void* d_ws, size_t ws_size,
                              hipStream_t stream) {
  const float* hs   = (const float*)d_in[0];
  const float* res  = (const float*)d_in[1];
  const float* ln1w = (const float*)d_in[2];
  const float* ln1b = (const float*)d_in[3];
  const float* wqkv = (const float*)d_in[4];
  const float* bqkv = (const float*)d_in[5];
  const float* wout = (const float*)d_in[6];
  const float* bout = (const float*)d_in[7];
  const float* ln2w = (const float*)d_in[8];
  const float* ln2b = (const float*)d_in[9];
  const float* wfc1 = (const float*)d_in[10];
  const float* bfc1 = (const float*)d_in[11];
  const float* wfc2 = (const float*)d_in[12];
  const float* bfc2 = (const float*)d_in[13];

  float* y      = (float*)d_out;                    // [B,S,E] fp32
  float* resid2 = y + (size_t)M * E;                // [B,S,E] fp32

  char* ws = (char*)d_ws;
  size_t off = 0;
  auto alloc = [&](size_t bytes) {
    size_t o = off;
    off += (bytes + 255) & ~(size_t)255;
    return o;
  };
  float*  resid   = (float*)(ws + alloc(sizeof(float) * M * E));
  __bf16* hbuf    = (__bf16*)(ws + alloc(sizeof(__bf16) * M * E));
  __bf16* wqkv_t  = (__bf16*)(ws + alloc(sizeof(__bf16) * (size_t)E * 3 * E));
  __bf16* wout_t  = (__bf16*)(ws + alloc(sizeof(__bf16) * (size_t)E * E));
  __bf16* wfc1_t  = (__bf16*)(ws + alloc(sizeof(__bf16) * (size_t)E * F));
  __bf16* wfc2_t  = (__bf16*)(ws + alloc(sizeof(__bf16) * (size_t)F * E));
  __bf16* qbuf    = (__bf16*)(ws + alloc(sizeof(__bf16) * (size_t)M * E));
  __bf16* kTbuf   = (__bf16*)(ws + alloc(sizeof(__bf16) * (size_t)M * E));
  __bf16* vbuf    = (__bf16*)(ws + alloc(sizeof(__bf16) * (size_t)M * E));
  __bf16* ctxbuf  = (__bf16*)(ws + alloc(sizeof(__bf16) * (size_t)M * E));
  __bf16* actbuf  = (__bf16*)(ws + alloc(sizeof(__bf16) * (size_t)M * F));
  (void)in_sizes; (void)n_in; (void)out_size; (void)ws_size;

  // 1) weight downconvert + transpose (fp32 [N,K] -> bf16 [K,N]), tiled
  wt_kernel<<<(3 * E / 32) * (E / 32), 256, 0, stream>>>(wqkv, wqkv_t, 3 * E, E);
  wt_kernel<<<(E / 32) * (E / 32),     256, 0, stream>>>(wout, wout_t, E, E);
  wt_kernel<<<(F / 32) * (E / 32),     256, 0, stream>>>(wfc1, wfc1_t, F, E);
  wt_kernel<<<(E / 32) * (F / 32),     256, 0, stream>>>(wfc2, wfc2_t, E, F);

  // 2) resid = hs + residual ; h = LN1(resid)
  add_ln_kernel<<<M, 256, 0, stream>>>(hs, res, ln1w, ln1b, resid, hbuf);

  // 3) QKV GEMM: [M,1024] x [1024,3072] -> scatter q / kT / v
  gemm_kernel<0><<<dim3(3 * E / 128, M / 128), 256, 0, stream>>>(
      hbuf, wqkv_t, bqkv, 3 * E, E, nullptr, nullptr, nullptr, qbuf, kTbuf, vbuf);

  // 4) causal flash attention -> ctx [M,E] bf16
  attn_kernel<<<dim3(S / 128, BB * H), 256, 0, stream>>>(qbuf, kTbuf, vbuf, ctxbuf);

  // 5) out projection + residual -> resid2 (fp32, second output)
  gemm_kernel<1><<<dim3(E / 128, M / 128), 256, 0, stream>>>(
      ctxbuf, wout_t, bout, E, E, resid2, nullptr, resid, nullptr, nullptr, nullptr);

  // 6) h2 = LN2(resid2) (bf16, reuse hbuf)
  add_ln_kernel<<<M, 256, 0, stream>>>(resid2, nullptr, ln2w, ln2b, nullptr, hbuf);

  // 7) fc1 + exact GELU -> act [M,F] bf16
  gemm_kernel<2><<<dim3(F / 128, M / 128), 256, 0, stream>>>(
      hbuf, wfc1_t, bfc1, F, E, nullptr, actbuf, nullptr, nullptr, nullptr, nullptr);

  // 8) fc2 -> y (fp32, first output)
  gemm_kernel<3><<<dim3(E / 128, M / 128), 256, 0, stream>>>(
      actbuf, wfc2_t, bfc2, E, F, y, nullptr, nullptr, nullptr, nullptr, nullptr);
}
